// AFM_65867618452110
// MI455X (gfx1250) — matrix operand order repaired and checked by reading, not compiled, not run on hardware
//
#include <hip/hip_runtime.h>

typedef __attribute__((ext_vector_type(16))) _Float16 v16h;
typedef __attribute__((ext_vector_type(8)))  _Float16 v8h;
typedef __attribute__((ext_vector_type(8)))  float    v8f;

#define NFIELD 26
#define EMBD   64
#define AFD    64
#define NPAIR  325     // 26*25/2
#define NPAIRP 336     // padded to 21 tiles of 16
#define NTILE  21
#define DDIM   13
#define EMBPAD 68      // floats per emb row: 16B-aligned, bank-rotated (+4)
#define W1PAD  72      // halves per W1^T row: 16B-aligned, bank-rotated
#define BLOCK  128

__global__ __launch_bounds__(BLOCK)
void afm_wmma_kernel(const int*   __restrict__ sparse,
                     const float* __restrict__ dense,
                     const float* __restrict__ table,
                     const float* __restrict__ lin,
                     const float* __restrict__ wd,
                     const float* __restrict__ bias,
                     const float* __restrict__ W1,
                     const float* __restrict__ b1,
                     const float* __restrict__ w2,
                     const float* __restrict__ pvec,
                     float* __restrict__ out)
{
    __shared__ float    s_emb[(NFIELD + 1) * EMBPAD];   // row 26 = zeros (pad pairs)
    __shared__ _Float16 s_W1t[AFD * W1PAD];             // W1 transposed: [a][k]
    __shared__ float    s_attw[NPAIRP];
    __shared__ int      s_pi[NPAIRP];
    __shared__ int      s_pj[NPAIRP];
    __shared__ float    s_part[4][32 * 8];              // per-wave rowsum staging
    __shared__ float    s_attout[8 * EMBD];
    __shared__ float    s_red[BLOCK];
    __shared__ float    s_lin;

    const int tid    = threadIdx.x;
    const int b      = blockIdx.x;
    const int lane   = tid & 31;
    const int wave   = tid >> 5;
    const int col    = lane & 15;   // N column within 16-tile / A row M
    const int laneHi = lane >> 4;   // lane group 0/1

    // ---- pair -> (i,j) tables; padded pairs point at the zero row ----
    for (int p0 = tid; p0 < NPAIRP; p0 += BLOCK) {
        int i = 0, rem = p0, cnt = NFIELD - 1;
        if (p0 < NPAIR) {
            while (rem >= cnt) { rem -= cnt; ++i; --cnt; }
            s_pi[p0] = i;
            s_pj[p0] = i + 1 + rem;
        } else {
            s_pi[p0] = NFIELD;      // zero row
            s_pj[p0] = NFIELD;
        }
    }
    // ---- embedding gather into LDS, float4 (table rows are 256B) ----
    for (int e = tid; e < NFIELD * 16; e += BLOCK) {
        int f = e >> 4, c4 = e & 15;
        const float4* src = (const float4*)(table + sparse[b * NFIELD + f] * EMBD);
        *(float4*)(s_emb + f * EMBPAD + c4 * 4) = src[c4];
    }
    // ---- zero row for padded pairs ----
    for (int e = tid; e < EMBD; e += BLOCK)
        s_emb[NFIELD * EMBPAD + e] = 0.f;
    // ---- W1 -> f16, transposed: s_W1t[a][k] ----
    for (int e = tid; e < EMBD * AFD; e += BLOCK) {
        int k = e >> 6, a = e & 63;
        s_W1t[a * W1PAD + k] = (_Float16)W1[e];
    }
    // ---- linear part partials ----
    float lp = 0.f;
    if (tid < NFIELD) lp += lin[sparse[b * NFIELD + tid]];
    if (tid < DDIM)   lp += dense[b * DDIM + tid] * wd[tid];
    s_red[tid] = lp;
    __syncthreads();
    for (int s = 64; s > 0; s >>= 1) {
        if (tid < s) s_red[tid] += s_red[tid + s];
        __syncthreads();
    }
    if (tid == 0) s_lin = s_red[0] + bias[0];

    // ---- per-lane bias/w2 for the 4 N-tiles ----
    float biasv[4], w2v[4];
#pragma unroll
    for (int n = 0; n < 4; ++n) { int a = n * 16 + col; biasv[n] = b1[a]; w2v[n] = w2[a]; }

    // ---- W1 B-fragments (once, kept in VGPRs), 2x b128 per fragment ----
    // B 32x16 16-bit layout: lanes 0-15 hold K=0..7 (h<8), K=16..23 (h>=8),
    // lanes 16-31 hold K=8..15, 24..31; column N = lane%16.
    v16h bfr[4][2];
#pragma unroll
    for (int n = 0; n < 4; ++n) {
        const v8h* wrow = (const v8h*)(s_W1t + (n * 16 + col) * W1PAD);
#pragma unroll
        for (int kk = 0; kk < 2; ++kk) {
            v8h lo = wrow[kk * 4 + laneHi];
            v8h hi = wrow[kk * 4 + laneHi + 2];
#pragma unroll
            for (int h = 0; h < 8; ++h) { bfr[n][kk][h] = lo[h]; bfr[n][kk][8 + h] = hi[h]; }
        }
    }

    // ---- pass 1: att_w via WMMA, tiles of 16 pairs, round-robin per wave ----
    for (int t = wave; t < NTILE; t += 4) {
        const int pr = t * 16 + col;            // A row M = lane%16
        const float4* ri4 = (const float4*)(s_emb + s_pi[pr] * EMBPAD);
        const float4* rj4 = (const float4*)(s_emb + s_pj[pr] * EMBPAD);
        // A 16x32 16-bit layout: lane<16: v0-3 K=0..7, v4-7 K=16..23; lane>=16: +8
        // -> 4 contiguous 8-float runs per source row, all b128 loads.
        v16h a0, a1;
#pragma unroll
        for (int r = 0; r < 4; ++r) {
            const int q = r * 4 + laneHi * 2;
            const float4 fi0 = ri4[q],     fj0 = rj4[q];
            const float4 fi1 = ri4[q + 1], fj1 = rj4[q + 1];
            v16h& dst = (r < 2) ? a0 : a1;
            const int o = (r & 1) * 8;
            dst[o + 0] = (_Float16)(fi0.x * fj0.x);
            dst[o + 1] = (_Float16)(fi0.y * fj0.y);
            dst[o + 2] = (_Float16)(fi0.z * fj0.z);
            dst[o + 3] = (_Float16)(fi0.w * fj0.w);
            dst[o + 4] = (_Float16)(fi1.x * fj1.x);
            dst[o + 5] = (_Float16)(fi1.y * fj1.y);
            dst[o + 6] = (_Float16)(fi1.z * fj1.z);
            dst[o + 7] = (_Float16)(fi1.w * fj1.w);
        }
        float rowsum[8] = {0.f,0.f,0.f,0.f,0.f,0.f,0.f,0.f};
#pragma unroll
        for (int n = 0; n < 4; ++n) {
            v8f c = {};
            c = __builtin_amdgcn_wmma_f32_16x16x32_f16(false, a0, false, bfr[n][0],
                                                       (short)0, c, false, false);
            c = __builtin_amdgcn_wmma_f32_16x16x32_f16(false, a1, false, bfr[n][1],
                                                       (short)0, c, false, false);
#pragma unroll
            for (int i = 0; i < 8; ++i) {
                float hv = c[i] + biasv[n];          // + b1
                hv = hv > 0.f ? hv : 0.f;            // ReLU
                rowsum[i] += hv * w2v[n];            // dot with w2 (partial over cols)
            }
        }
        // Column reduction via per-wave LDS staging (in-order DS within a wave):
        // lane srcLane holds rows (srcLane>=16)*8+i at column srcLane&15.
        float4* sp = (float4*)&s_part[wave][lane * 8];
        float4 r0 = make_float4(rowsum[0], rowsum[1], rowsum[2], rowsum[3]);
        float4 r1 = make_float4(rowsum[4], rowsum[5], rowsum[6], rowsum[7]);
        sp[0] = r0;
        sp[1] = r1;
        const int row = lane & 15;      // each row handled by 2 lanes (col halves)
        const float* basep = &s_part[wave][(((row >> 3) * 16) + (laneHi * 8)) * 8 + (row & 7)];
        float v = 0.f;
#pragma unroll
        for (int cc = 0; cc < 8; ++cc) v += basep[cc * 8];
        v += __shfl_xor(v, 16, 32);     // merge the two column halves
        if (lane < 16) s_attw[t * 16 + row] = v;
    }
    __syncthreads();

    // ---- softmax over 325 pairs ----
    float mx = -3.0e38f;
    for (int p0 = tid; p0 < NPAIR; p0 += BLOCK) mx = fmaxf(mx, s_attw[p0]);
    s_red[tid] = mx;
    __syncthreads();
    for (int s = 64; s > 0; s >>= 1) {
        if (tid < s) s_red[tid] = fmaxf(s_red[tid], s_red[tid + s]);
        __syncthreads();
    }
    const float gmx = s_red[0];
    __syncthreads();
    float sm = 0.f;
    for (int p0 = tid; p0 < NPAIR; p0 += BLOCK) {
        float e = __expf(s_attw[p0] - gmx);
        s_attw[p0] = e;
        sm += e;
    }
    s_red[tid] = sm;
    __syncthreads();
    for (int s = 64; s > 0; s >>= 1) {
        if (tid < s) s_red[tid] += s_red[tid + s];
        __syncthreads();
    }
    const float den = s_red[0];

    // ---- pass 2: score-weighted sum of att_x (fp32, float4-vectorized) ----
    const int d4 = (tid & 15) * 4;      // 4 dims per thread
    const int hh = tid >> 4;            // 8 pair-strides
    float4 acc4 = make_float4(0.f, 0.f, 0.f, 0.f);
    for (int p0 = hh; p0 < NPAIR; p0 += 8) {
        const float sc = s_attw[p0];
        const float4 a  = *(const float4*)(s_emb + s_pi[p0] * EMBPAD + d4);
        const float4 bb = *(const float4*)(s_emb + s_pj[p0] * EMBPAD + d4);
        acc4.x += sc * a.x * bb.x;
        acc4.y += sc * a.y * bb.y;
        acc4.z += sc * a.z * bb.z;
        acc4.w += sc * a.w * bb.w;
    }
    *(float4*)(s_attout + hh * EMBD + d4) = acc4;
    __syncthreads();
    float fin = 0.f;
    if (tid < EMBD) {
        float s = 0.f;
#pragma unroll
        for (int q = 0; q < 8; ++q) s += s_attout[q * EMBD + tid];
        fin = s * pvec[tid];
    }
    s_red[tid] = fin;
    __syncthreads();
    for (int s = 64; s > 0; s >>= 1) {
        if (tid < s) s_red[tid] += s_red[tid + s];
        __syncthreads();
    }
    if (tid == 0) out[b] = s_lin + s_red[0] / den;
}

extern "C" void kernel_launch(void* const* d_in, const int* in_sizes, int n_in,
                              void* d_out, int out_size, void* d_ws, size_t ws_size,
                              hipStream_t stream) {
    const int*   sparse = (const int*)  d_in[0];
    const float* dense  = (const float*)d_in[1];
    const float* table  = (const float*)d_in[2];
    const float* lin    = (const float*)d_in[3];
    const float* wd     = (const float*)d_in[4];
    const float* bias   = (const float*)d_in[5];
    const float* W1     = (const float*)d_in[6];
    const float* b1     = (const float*)d_in[7];
    const float* w2     = (const float*)d_in[8];
    const float* pvec   = (const float*)d_in[9];
    float* outp = (float*)d_out;

    const int B = in_sizes[0] / NFIELD;   // 4096
    afm_wmma_kernel<<<B, BLOCK, 0, stream>>>(sparse, dense, table, lin, wd, bias,
                                             W1, b1, w2, pvec, outp);
}
